// FDNet_11673721110696
// MI455X (gfx1250) — matrix operand order repaired and checked by reading, not compiled, not run on hardware
//
#include <hip/hip_runtime.h>
#include <hip/hip_bf16.h>

// ---------------------------------------------------------------------------
// FDNet pointer-decoder for MI455X (gfx1250, wave32, WMMA bf16 16x16x32)
// ---------------------------------------------------------------------------
#define Bsz 4096
#define Nn  50
#define Dd  128
#define DV  8
#define FS  4
#define SE  20
#define TPH 64
#define NEGV (-1e9f)

typedef __attribute__((ext_vector_type(16))) __bf16 v16bf;
typedef __attribute__((ext_vector_type(8)))  __bf16 v8bf;
typedef __attribute__((ext_vector_type(8)))  float  v8f;
typedef __attribute__((ext_vector_type(4)))  float  v4f;

__device__ __forceinline__ float sigmf(float v) { return 1.f / (1.f + __expf(-v)); }

// convert 8 contiguous f32 -> v8bf
__device__ __forceinline__ v8bf cvt8(const float* __restrict__ src) {
  v4f f0 = *(const v4f*)src, f1 = *(const v4f*)(src + 4);
  v8bf o;
#pragma unroll
  for (int i = 0; i < 4; ++i) { o[i] = (__bf16)f0[i]; o[4 + i] = (__bf16)f1[i]; }
  return o;
}

// build v16bf fragment from two contiguous 16-byte LDS chunks
__device__ __forceinline__ v16bf frag16(const __bf16* __restrict__ p0,
                                        const __bf16* __restrict__ p1) {
  v8bf lo = *(const v8bf*)p0, hi = *(const v8bf*)p1;
  return __builtin_shufflevector(lo, hi, 0, 1, 2, 3, 4, 5, 6, 7,
                                 8, 9, 10, 11, 12, 13, 14, 15);
}

// ======================= one-time init / weight convert =====================
__global__ void init_state_kernel(const float* __restrict__ h0,
                                  const float* __restrict__ c0,
                                  const float* __restrict__ dec,
                                  const unsigned char* __restrict__ vrm,
                                  const int* __restrict__ start_idx,
                                  float* __restrict__ h, float* __restrict__ c,
                                  float* __restrict__ x,
                                  unsigned char* __restrict__ mask,
                                  int* __restrict__ prev, int* __restrict__ last) {
  size_t i = (size_t)blockIdx.x * 256 + threadIdx.x;
  if (i < (size_t)Bsz * Dd) { h[i] = h0[i]; c[i] = c0[i]; x[i] = dec[i]; }
  if (i < (size_t)Bsz * Nn) mask[i] = vrm[i] ? 1 : 0;
  if (i < Bsz) { prev[i] = 0; last[i] = start_idx[i]; }
}

__global__ void cvt_weights_kernel(const float* __restrict__ Wih, const float* __restrict__ Whh,
                                   const float* __restrict__ gWq, const float* __restrict__ pWq,
                                   const float* __restrict__ gWr, const float* __restrict__ pWr,
                                   __bf16* o_ih, __bf16* o_hh, __bf16* o_gq,
                                   __bf16* o_pq, __bf16* o_gr, __bf16* o_pr) {
  int i = blockIdx.x * 256 + threadIdx.x;
  if (i < 4 * Dd * Dd) { o_ih[i] = (__bf16)Wih[i]; o_hh[i] = (__bf16)Whh[i]; }
  if (i < Dd * Dd) {
    o_gq[i] = (__bf16)gWq[i]; o_pq[i] = (__bf16)pWq[i];
    o_gr[i] = (__bf16)gWr[i]; o_pr[i] = (__bf16)pWr[i];
  }
}

// ======================= WMMA GEMM: Y[M,128] = A[M,128] * W^T + bias ========
// A f32 (staged to bf16 via LDS), W bf16 row-major [128,128] ([n][k] in LDS,
// k-contiguous, stride 40 -> B-fragment = two ds_load_b128 per lane).
template <typename OutT>
__global__ void gemm128_kernel(const float* __restrict__ A,
                               const __bf16* __restrict__ Wbf,
                               const float* __restrict__ bias,
                               OutT* __restrict__ out, int M) {
  __shared__ __bf16 Al[64 * 40];    // 64 rows x 32 k (stride 40)
  __shared__ __bf16 Wl[128 * 40];   // 128 n x 32 k  (stride 40)
  const int tid = threadIdx.x, wave = tid >> 5, lane = tid & 31;
  const int rowBase = blockIdx.x * 64;
  const int rb = wave & 3, ch = wave >> 2;     // row band, column half
  const int r16 = lane & 15, half = lane >> 4;
  v8f acc[4] = {};
  for (int kc = 0; kc < 4; ++kc) {
    const int k0 = kc * 32;
    __syncthreads();
    {  // stage A: 2048 elems = 256 chunks of 8 -> one per thread
      const int e = tid * 8, r = e >> 5, k = e & 31;
      *(v8bf*)&Al[r * 40 + k] = cvt8(&A[(size_t)(rowBase + r) * Dd + k0 + k]);
    }
#pragma unroll
    for (int c8 = 0; c8 < 2; ++c8) {  // stage W rows: 4096 elems = 512 chunks
      const int e = (tid + c8 * 256) * 8, n = e >> 5, k = e & 31;
      *(v8bf*)&Wl[n * 40 + k] = *(const v8bf*)&Wbf[(size_t)n * Dd + k0 + k];
    }
    if (kc < 3)  // gfx1250 global_prefetch_b8 for next A chunk
      __builtin_prefetch(&A[(size_t)(rowBase + (tid >> 2)) * Dd + k0 + 32], 0, 1);
    __syncthreads();
    const __bf16* Ar = &Al[(rb * 16 + r16) * 40 + half * 8];
    const v16bf afrag = frag16(Ar, Ar + 16);   // ISA 16-bit A 16x32 layout
#pragma unroll
    for (int ct = 0; ct < 4; ++ct) {
      const int col = ch * 64 + ct * 16 + r16;
      const __bf16* Br = &Wl[col * 40 + half * 8];
      const v16bf bfrag = frag16(Br, Br + 16);
      acc[ct] = __builtin_amdgcn_wmma_f32_16x16x32_bf16(
          false, afrag, false, bfrag, (short)0, acc[ct], false, false);
    }
  }
#pragma unroll
  for (int ct = 0; ct < 4; ++ct) {
    const int col = ch * 64 + ct * 16 + r16;
    const float bv = bias[col];
#pragma unroll
    for (int j = 0; j < 8; ++j) {
      const int row = rowBase + rb * 16 + half * 8 + j;
      out[(size_t)row * Dd + col] = (OutT)(acc[ct][j] + bv);
    }
  }
}

// ======================= fused WMMA LSTM cell ===============================
// gates[b,0:512] = x*Wih^T + h*Whh^T + b ; then h,c update in place.
// Block: 16 rows x 512 cols; 8 waves x 4 col-tiles; K = 256 (x then h).
__global__ void lstm_kernel(const float* __restrict__ x,
                            float* __restrict__ h, float* __restrict__ c,
                            const __bf16* __restrict__ Wih,
                            const __bf16* __restrict__ Whh,
                            const float* __restrict__ bih,
                            const float* __restrict__ bhh) {
  __shared__ __attribute__((aligned(16))) char smem[2048 + 512 * 40 * 2];
  __bf16* Al = (__bf16*)smem;                // 16x32 (stride 40) = 1280 B
  __bf16* Wl = (__bf16*)(smem + 2048);       // 512 n x 32 k (stride 40) = 40960 B
  float*  Gl = (float*)smem;                 // reused: 16x528 f32 = 33792 B
  const int tid = threadIdx.x, wave = tid >> 5, lane = tid & 31;
  const int rowBase = blockIdx.x * 16;
  const int r16 = lane & 15, half = lane >> 4;
  v8f acc[4] = {};
  for (int kc = 0; kc < 8; ++kc) {
    const float*  Asrc = (kc < 4) ? x : h;
    const __bf16* Wsrc = (kc < 4) ? Wih : Whh;
    const int k0 = (kc & 3) * 32;
    __syncthreads();
    if (tid < 64) {  // stage A: 512 elems = 64 chunks of 8
      const int e = tid * 8, r = e >> 5, k = e & 31;
      *(v8bf*)&Al[r * 40 + k] = cvt8(&Asrc[(size_t)(rowBase + r) * Dd + k0 + k]);
    }
#pragma unroll
    for (int c8 = 0; c8 < 8; ++c8) {  // stage W rows: 16384 elems = 2048 chunks
      const int e = (tid + c8 * 256) * 8, j = e >> 5, k = e & 31;
      *(v8bf*)&Wl[j * 40 + k] = *(const v8bf*)&Wsrc[(size_t)j * Dd + k0 + k];
    }
    __syncthreads();
    const __bf16* Ar = &Al[r16 * 40 + half * 8];
    const v16bf afrag = frag16(Ar, Ar + 16);
#pragma unroll
    for (int ct = 0; ct < 4; ++ct) {
      const int col = (wave * 4 + ct) * 16 + r16;
      const __bf16* Br = &Wl[col * 40 + half * 8];
      const v16bf bfrag = frag16(Br, Br + 16);
      acc[ct] = __builtin_amdgcn_wmma_f32_16x16x32_bf16(
          false, afrag, false, bfrag, (short)0, acc[ct], false, false);
    }
  }
  __syncthreads();   // done with Wl/Al; reuse LDS for gates
#pragma unroll
  for (int ct = 0; ct < 4; ++ct) {
    const int col0 = (wave * 4 + ct) * 16 + r16;
#pragma unroll
    for (int j = 0; j < 8; ++j) Gl[(half * 8 + j) * 528 + col0] = acc[ct][j];
  }
  __syncthreads();
  for (int idx = tid; idx < 16 * Dd; idx += 256) {
    const int d = idx & 127, r = idx >> 7;
    const size_t g = (size_t)r * 528;
    const float gi = Gl[g + d]        + bih[d]        + bhh[d];
    const float gf = Gl[g + 128 + d]  + bih[128 + d]  + bhh[128 + d];
    const float gg = Gl[g + 256 + d]  + bih[256 + d]  + bhh[256 + d];
    const float go = Gl[g + 384 + d]  + bih[384 + d]  + bhh[384 + d];
    const size_t off = (size_t)(rowBase + r) * Dd + d;
    const float c2 = sigmf(gf) * c[off] + sigmf(gi) * tanhf(gg);
    c[off] = c2;
    h[off] = sigmf(go) * tanhf(c2);
  }
}

// ======================= fused glimpse attention (block per b) ==============
__global__ void glimpse_kernel(const float* __restrict__ qg,
                               const __bf16* __restrict__ Eg,
                               const float* __restrict__ gv,
                               unsigned char* __restrict__ mask,
                               const int* __restrict__ prev,
                               float* __restrict__ gl, int t) {
  __shared__ __bf16 El[Nn * Dd];
  __shared__ float qls[Dd], vls[Dd], sArr[Nn], wArr[Nn];
  const int b = blockIdx.x, tid = threadIdx.x, wave = tid >> 5, lane = tid & 31;
  if (tid == 0) {   // mask carry update (scatter prev at t>0, then mask_modify)
    unsigned char* mrow = mask + (size_t)b * Nn;
    if (t > 0) mrow[prev[b]] = 1;
    bool all = true;
    for (int n = 0; n < Nn; ++n) all = all && (mrow[n] != 0);
    if (all) mrow[Nn - 1] = 0;
  }
  for (int c8 = tid; c8 < Nn * Dd / 8; c8 += 256) {   // 16-byte tile copies
    const int e = c8 * 8, n = e >> 7, d = e & 127;
    *(v8bf*)&El[e] = *(const v8bf*)&Eg[((size_t)n * Bsz + b) * Dd + d];
  }
  if (tid < Dd) { qls[tid] = qg[(size_t)b * Dd + tid]; vls[tid] = gv[tid]; }
  __syncthreads();
  for (int n = wave; n < Nn; n += 8) {
    float s = 0.f;
    for (int d = lane; d < Dd; d += 32)
      s += vls[d] * tanhf(qls[d] + (float)El[n * Dd + d]);
    for (int off = 16; off; off >>= 1) s += __shfl_xor(s, off, 32);
    if (lane == 0) sArr[n] = mask[(size_t)b * Nn + n] ? NEGV : s;
  }
  __syncthreads();
  if (tid == 0) {
    float m = -3.4e38f;
    for (int n = 0; n < Nn; ++n) m = fmaxf(m, sArr[n]);
    float sum = 0.f;
    for (int n = 0; n < Nn; ++n) { wArr[n] = __expf(sArr[n] - m); sum += wArr[n]; }
    const float inv = 1.f / sum;
    for (int n = 0; n < Nn; ++n) wArr[n] *= inv;
  }
  __syncthreads();
  if (tid < Dd) {
    float a = 0.f;
    for (int n = 0; n < Nn; ++n) a += wArr[n] * (float)El[n * Dd + tid];
    gl[(size_t)b * Dd + tid] = a;
  }
}

// ============ fused pointer + log_softmax + argmax + MLP + gather ===========
__global__ void pointer_kernel(const float* __restrict__ qp,
                               const __bf16* __restrict__ Ep,
                               const float* __restrict__ pv,
                               const unsigned char* __restrict__ mask,
                               const float* __restrict__ cu,
                               const float* __restrict__ Vfeat,
                               const float* __restrict__ start_fea,
                               const float* __restrict__ step_table,
                               const float* __restrict__ W1, const float* __restrict__ b1,
                               const float* __restrict__ W2, const float* __restrict__ b2,
                               float* __restrict__ x,
                               int* __restrict__ prev, int* __restrict__ last,
                               float* __restrict__ out, int t) {
  __shared__ __bf16 El[Nn * Dd];
  __shared__ float qls[Dd], vls[Dd], lArr[Nn], red[TPH];
  __shared__ float lse;
  __shared__ int selIdx;
  const int b = blockIdx.x, tid = threadIdx.x, wave = tid >> 5, lane = tid & 31;
  for (int c8 = tid; c8 < Nn * Dd / 8; c8 += 256) {
    const int e = c8 * 8, n = e >> 7, d = e & 127;
    *(v8bf*)&El[e] = *(const v8bf*)&Ep[((size_t)n * Bsz + b) * Dd + d];
  }
  if (tid < Dd) { qls[tid] = qp[(size_t)b * Dd + tid]; vls[tid] = pv[tid]; }
  __syncthreads();
  for (int n = wave; n < Nn; n += 8) {
    float s = 0.f;
    for (int d = lane; d < Dd; d += 32)
      s += vls[d] * tanhf(qls[d] + (float)El[n * Dd + d]);
    for (int off = 16; off; off >>= 1) s += __shfl_xor(s, off, 32);
    if (lane == 0)
      lArr[n] = mask[(size_t)b * Nn + n] ? NEGV : 10.f * tanhf(s);
  }
  __syncthreads();
  if (tid == 0) {
    float m = -3.4e38f; int am = 0;
    for (int n = 0; n < Nn; ++n) if (lArr[n] > m) { m = lArr[n]; am = n; }
    float sum = 0.f;
    for (int n = 0; n < Nn; ++n) sum += __expf(lArr[n] - m);
    lse = m + __logf(sum);
    selIdx = am;
  }
  const int lastIdx = last[b];     // read before tid0 overwrites (post-barrier)
  __syncthreads();
  const int idx = selIdx;
  if (tid < Nn) out[((size_t)b * Nn + t) * Nn + tid] = lArr[tid] - lse;
  if (tid < TPH) {                 // 40 -> 64 -> 1 time-prediction MLP
    float in[2 * DV + FS + SE];
    for (int k = 0; k < DV; ++k) in[k]      = Vfeat[((size_t)b * Nn + lastIdx) * DV + k];
    for (int k = 0; k < DV; ++k) in[DV + k] = Vfeat[((size_t)b * Nn + idx) * DV + k];
    for (int f = 0; f < FS; ++f) in[2 * DV + f] = start_fea[(size_t)b * FS + f];
    for (int s = 0; s < SE; ++s) in[2 * DV + FS + s] = step_table[t * SE + s];
    float a = b1[tid];
    for (int q = 0; q < 2 * DV + FS + SE; ++q) a += in[q] * W1[q * TPH + tid];
    red[tid] = fmaxf(a, 0.f) * W2[tid];
  }
  if (tid < Dd)                    // next decoder input: context gather
    x[(size_t)b * Dd + tid] = cu[((size_t)idx * Bsz + b) * Dd + tid];
  __syncthreads();
  if (tid == 0) {
    float p = b2[0];
    for (int j = 0; j < TPH; ++j) p += red[j];
    const size_t LP = (size_t)Bsz * Nn * Nn;
    out[LP + (size_t)b * Nn + t] = (float)idx;                         // selections
    out[LP + (size_t)Bsz * Nn + (size_t)b * Nn + t] = p;               // pred_t
    out[LP + (size_t)2 * Bsz * Nn + (size_t)b * Nn + t] = p;           // pred_eta
    prev[b] = idx; last[b] = idx;
  }
}

// ======================= host-side orchestration ============================
extern "C" void kernel_launch(void* const* d_in, const int* in_sizes, int n_in,
                              void* d_out, int out_size, void* d_ws, size_t ws_size,
                              hipStream_t stream) {
  (void)in_sizes; (void)n_in; (void)out_size; (void)ws_size;
  const float* dec   = (const float*)d_in[0];
  const float* h0    = (const float*)d_in[1];
  const float* c0    = (const float*)d_in[2];
  const unsigned char* vrm = (const unsigned char*)d_in[3];
  const int*   sidx  = (const int*)d_in[4];
  const float* Vfeat = (const float*)d_in[5];
  const float* cu    = (const float*)d_in[8];
  const float* sfea  = (const float*)d_in[10];
  const float* Wih   = (const float*)d_in[11];
  const float* Whh   = (const float*)d_in[12];
  const float* bih   = (const float*)d_in[13];
  const float* bhh   = (const float*)d_in[14];
  const float* gWq   = (const float*)d_in[15];
  const float* gbq   = (const float*)d_in[16];
  const float* gWr   = (const float*)d_in[17];
  const float* gbr   = (const float*)d_in[18];
  const float* gv    = (const float*)d_in[19];
  const float* pWq   = (const float*)d_in[20];
  const float* pbq   = (const float*)d_in[21];
  const float* pWr   = (const float*)d_in[22];
  const float* pbr   = (const float*)d_in[23];
  const float* pv    = (const float*)d_in[24];
  const float* stbl  = (const float*)d_in[25];
  const float* tpW1  = (const float*)d_in[26];
  const float* tpb1  = (const float*)d_in[27];
  const float* tpW2  = (const float*)d_in[28];
  const float* tpb2  = (const float*)d_in[29];

  char* w = (char*)d_ws;
  auto alloc = [&](size_t bytes) -> char* {
    char* p = w; w += (bytes + 255) & ~(size_t)255; return p;
  };
  const size_t EGN = (size_t)Nn * Bsz * Dd;
  __bf16* Eg     = (__bf16*)alloc(EGN * 2);
  __bf16* Ep     = (__bf16*)alloc(EGN * 2);
  __bf16* Wih_bf = (__bf16*)alloc((size_t)4 * Dd * Dd * 2);
  __bf16* Whh_bf = (__bf16*)alloc((size_t)4 * Dd * Dd * 2);
  __bf16* gWq_bf = (__bf16*)alloc((size_t)Dd * Dd * 2);
  __bf16* pWq_bf = (__bf16*)alloc((size_t)Dd * Dd * 2);
  __bf16* gWr_bf = (__bf16*)alloc((size_t)Dd * Dd * 2);
  __bf16* pWr_bf = (__bf16*)alloc((size_t)Dd * Dd * 2);
  float* h  = (float*)alloc((size_t)Bsz * Dd * 4);
  float* c  = (float*)alloc((size_t)Bsz * Dd * 4);
  float* x  = (float*)alloc((size_t)Bsz * Dd * 4);
  float* qg = (float*)alloc((size_t)Bsz * Dd * 4);
  float* qp = (float*)alloc((size_t)Bsz * Dd * 4);
  float* gl = (float*)alloc((size_t)Bsz * Dd * 4);
  unsigned char* mask = (unsigned char*)alloc((size_t)Bsz * Nn);
  int* prev = (int*)alloc((size_t)Bsz * 4);
  int* last = (int*)alloc((size_t)Bsz * 4);

  // one-time setup
  init_state_kernel<<<(Bsz * Dd + 255) / 256, 256, 0, stream>>>(
      h0, c0, dec, vrm, sidx, h, c, x, mask, prev, last);
  cvt_weights_kernel<<<(4 * Dd * Dd + 255) / 256, 256, 0, stream>>>(
      Wih, Whh, gWq, pWq, gWr, pWr, Wih_bf, Whh_bf, gWq_bf, pWq_bf, gWr_bf, pWr_bf);
  // project_ref on fixed context: Eg/Ep = cu * W^T + b  (M = N*B = 204800)
  gemm128_kernel<__bf16><<<(Nn * Bsz) / 64, 256, 0, stream>>>(cu, gWr_bf, gbr, Eg, Nn * Bsz);
  gemm128_kernel<__bf16><<<(Nn * Bsz) / 64, 256, 0, stream>>>(cu, pWr_bf, pbr, Ep, Nn * Bsz);

  // sequential decode
  for (int t = 0; t < Nn; ++t) {
    lstm_kernel<<<Bsz / 16, 256, 0, stream>>>(x, h, c, Wih_bf, Whh_bf, bih, bhh);
    gemm128_kernel<float><<<Bsz / 64, 256, 0, stream>>>(h, gWq_bf, gbq, qg, Bsz);
    glimpse_kernel<<<Bsz, 256, 0, stream>>>(qg, Eg, gv, mask, prev, gl, t);
    gemm128_kernel<float><<<Bsz / 64, 256, 0, stream>>>(gl, pWq_bf, pbq, qp, Bsz);
    pointer_kernel<<<Bsz, 256, 0, stream>>>(qp, Ep, pv, mask, cu, Vfeat, sfea, stbl,
                                            tpW1, tpb1, tpW2, tpb2,
                                            x, prev, last, (float*)d_out, t);
  }
}
